// CBFNet_51376398795110
// MI455X (gfx1250) — matrix operand order repaired and checked by reading, not compiled, not run on hardware
//
#include <hip/hip_runtime.h>
#include <hip/hip_bf16.h>
#include <math.h>

// ---------------------------------------------------------------------------
// Types for CDNA5 WMMA (wave32): 16x16x32 bf16 -> f32
// ---------------------------------------------------------------------------
typedef __attribute__((ext_vector_type(16))) __bf16       v16bf;
typedef __attribute__((ext_vector_type(8)))  float        v8f;
typedef __attribute__((ext_vector_type(8)))  unsigned int v8u;

#define ND   64
#define ED   32
#define MSGD 128
#define HIDD 256
#define KIN  160   // 2*ND + ED

__device__ __forceinline__ unsigned short f2bf(float f) {
  union { float f; unsigned u; } v; v.f = f;
  unsigned r = v.u + 0x7fffu + ((v.u >> 16) & 1u);   // round to nearest even
  return (unsigned short)(r >> 16);
}
__device__ __forceinline__ float bf2f(unsigned short u) {
  union { unsigned u; float f; } v; v.u = ((unsigned)u) << 16;
  return v.f;
}
// monotone float <-> int encoding so atomicMax(int) == float max
__device__ __forceinline__ int orderedEnc(float f) {
  int i = __float_as_int(f);
  return i >= 0 ? i : (i ^ 0x7fffffff);
}
__device__ __forceinline__ float orderedDec(int i) {
  return __int_as_float(i >= 0 ? i : (i ^ 0x7fffffff));
}

// A fragment: 16x32 bf16, row-major LDS tile [16][strideK].
// Layout used here (contiguous K, same as fragB): lane holds row M = lane&15; lanes 0-15: K 0-15,
// lanes 16-31: K 16-31; element pair (2v,2v+1) = consecutive K.
__device__ __forceinline__ v16bf fragA(const unsigned short* __restrict__ s,
                                       int strideK, int kbase, int lane) {
  const int row = lane & 15, half = lane >> 4;
  const unsigned short* p = s + row * strideK + kbase;
  v8u raw;
#pragma unroll
  for (int v = 0; v < 8; ++v) {
    int j = 2 * v;
    int K = j + 16 * half;
    raw[v] = *(const unsigned int*)(p + K);
  }
  return __builtin_bit_cast(v16bf, raw);
}

// B fragment: 32x16 bf16. Weights staged in LDS transposed as [N][K] so
// consecutive K are contiguous. lane col N = lane&15; lanes 0-15 K=0-15,
// lanes 16-31 K=16-31 (element j -> K = j + 16*half).
__device__ __forceinline__ v16bf fragB(const unsigned short* __restrict__ s,
                                       int strideK, int n0, int kbase, int lane) {
  const int col = lane & 15, half = lane >> 4;
  const unsigned short* p = s + (n0 + col) * strideK + kbase + 16 * half;
  v8u raw;
#pragma unroll
  for (int v = 0; v < 8; ++v) raw[v] = *(const unsigned int*)(p + 2 * v);
  return __builtin_bit_cast(v16bf, raw);
}

__device__ __forceinline__ v8f wmma_bf16(v16bf a, v16bf b, v8f c) {
  return __builtin_amdgcn_wmma_f32_16x16x32_bf16(false, a, false, b,
                                                 (short)0, c, false, false);
}

// Intra-wave LDS RAW fence (CDNA5 split counters); also a compiler barrier.
__device__ __forceinline__ void wave_lds_fence() {
  asm volatile("s_wait_dscnt 0" ::: "memory");
}

// ---------------------------------------------------------------------------
// Init workspace: segmax = enc(-inf), denom = 0, aggr = 0
// ---------------------------------------------------------------------------
__global__ void cbf_init_kernel(int* segMax, float* denom, float* aggr, int nNodes) {
  int total = nNodes * MSGD;
  for (int i = blockIdx.x * blockDim.x + threadIdx.x; i < total;
       i += gridDim.x * blockDim.x) {
    aggr[i] = 0.f;
    if (i < nNodes) { segMax[i] = (int)0x807fffffu; denom[i] = 0.f; }
  }
}

// ---------------------------------------------------------------------------
// K1: fused edge MLP (two WMMA GEMMs) + gate logits + segment max
// block = 128 threads = 4 waves; each wave owns a 16-edge tile
// ---------------------------------------------------------------------------
#define K1_PER_WAVE ((16*KIN + 16*HIDD)*2 + 16*MSGD*4)  // 21504 B
#define K1_SMEM (HIDD*KIN*2 + MSGD*HIDD*2 + (HIDD+MSGD+MSGD)*4 + 4*K1_PER_WAVE)

__global__ void __launch_bounds__(128)
cbf_edge_mlp_kernel(const float* __restrict__ nodeF, const float* __restrict__ edgeF,
                    const float* __restrict__ W1, const float* __restrict__ B1,
                    const float* __restrict__ W2, const float* __restrict__ B2,
                    const float* __restrict__ wGate, const float* __restrict__ bGate,
                    const int* __restrict__ senders, const int* __restrict__ receivers,
                    int nEdges,
                    unsigned short* __restrict__ gMsg, float* __restrict__ gLogits,
                    int* __restrict__ segMax) {
  extern __shared__ char smem[];
  unsigned short* sW1 = (unsigned short*)smem;           // [256][160] bf16
  unsigned short* sW2 = sW1 + HIDD * KIN;                // [128][256] bf16
  float* sB1   = (float*)(sW2 + MSGD * HIDD);            // 256
  float* sB2   = sB1 + HIDD;                             // 128
  float* sGate = sB2 + MSGD;                             // 128
  char*  pw    = (char*)(sGate + MSGD);

  const int wave = threadIdx.x >> 5, lane = threadIdx.x & 31;
  unsigned short* sA   = (unsigned short*)(pw + wave * K1_PER_WAVE); // [16][160]
  unsigned short* sH   = sA + 16 * KIN;                              // [16][256]
  float*          sMsg = (float*)(sH + 16 * HIDD);                   // [16][128]

  // Stage weights (transposed [N][K]) + biases into LDS
  for (int i = threadIdx.x; i < KIN * HIDD; i += blockDim.x) {
    int k = i / HIDD, n = i - k * HIDD;
    sW1[n * KIN + k] = f2bf(W1[i]);
  }
  for (int i = threadIdx.x; i < HIDD * MSGD; i += blockDim.x) {
    int k = i / MSGD, n = i - k * MSGD;
    sW2[n * HIDD + k] = f2bf(W2[i]);
  }
  for (int i = threadIdx.x; i < HIDD; i += blockDim.x) sB1[i] = B1[i];
  for (int i = threadIdx.x; i < MSGD; i += blockDim.x) { sB2[i] = B2[i]; sGate[i] = wGate[i]; }
  __syncthreads();

  const int col = lane & 15, half = lane >> 4;
  const int nTiles = (nEdges + 15) >> 4;

  for (int tile = blockIdx.x * 4 + wave; tile < nTiles; tile += gridDim.x * 4) {
    const int e0 = tile << 4;

    // Gather concat(sender_feats, receiver_feats, edge_feats) -> bf16 LDS
    for (int idx = lane; idx < 16 * KIN; idx += 32) {
      int r = idx / KIN, c = idx - r * KIN;
      int e = e0 + r;
      float val = 0.f;
      if (e < nEdges) {
        if (c < ND)            val = nodeF[(size_t)senders[e] * ND + c];
        else if (c < 2 * ND)   val = nodeF[(size_t)receivers[e] * ND + (c - ND)];
        else                   val = edgeF[(size_t)e * ED + (c - 2 * ND)];
      }
      sA[idx] = f2bf(val);
    }
    wave_lds_fence();

    // GEMM1: [16x160] @ [160x256] -> relu -> bf16 h in LDS
    for (int nt = 0; nt < 16; ++nt) {
      v8f acc = {};
#pragma unroll
      for (int kt = 0; kt < 5; ++kt)
        acc = wmma_bf16(fragA(sA, KIN, kt * 32, lane),
                        fragB(sW1, KIN, nt * 16, kt * 32, lane), acc);
      float bias = sB1[nt * 16 + col];
#pragma unroll
      for (int r = 0; r < 8; ++r) {
        float v = acc[r] + bias; v = v > 0.f ? v : 0.f;
        sH[(r + 8 * half) * HIDD + nt * 16 + col] = f2bf(v);
      }
    }
    wave_lds_fence();

    // GEMM2: [16x256] @ [256x128] -> relu -> msg (bf16 global + f32 LDS)
    for (int nt = 0; nt < 8; ++nt) {
      v8f acc = {};
#pragma unroll
      for (int kt = 0; kt < 8; ++kt)
        acc = wmma_bf16(fragA(sH, HIDD, kt * 32, lane),
                        fragB(sW2, HIDD, nt * 16, kt * 32, lane), acc);
      float bias = sB2[nt * 16 + col];
#pragma unroll
      for (int r = 0; r < 8; ++r) {
        float v = acc[r] + bias; v = v > 0.f ? v : 0.f;
        int row = r + 8 * half;
        sMsg[row * MSGD + nt * 16 + col] = v;
        int e = e0 + row;
        if (e < nEdges) gMsg[(size_t)e * MSGD + nt * 16 + col] = f2bf(v);
      }
    }
    wave_lds_fence();

    // Gate: logits + ordered-int segment max over receivers
    if (lane < 16) {
      int e = e0 + lane;
      if (e < nEdges) {
        float acc = bGate[0];
#pragma unroll 4
        for (int c = 0; c < MSGD; ++c) acc += sMsg[lane * MSGD + c] * sGate[c];
        gLogits[e] = acc;
        atomicMax(&segMax[receivers[e]], orderedEnc(acc));
      }
    }
  }
}

// ---------------------------------------------------------------------------
// K2: denom[r] = sum exp(logit - segmax[r])
// ---------------------------------------------------------------------------
__global__ void cbf_denom_kernel(const float* __restrict__ gLogits,
                                 const int* __restrict__ receivers,
                                 const int* __restrict__ segMax,
                                 float* __restrict__ denom, int nEdges) {
  int e = blockIdx.x * blockDim.x + threadIdx.x;
  if (e >= nEdges) return;
  int r = receivers[e];
  float ex = __expf(gLogits[e] - orderedDec(segMax[r]));
  atomicAdd(&denom[r], ex);
}

// ---------------------------------------------------------------------------
// K3: aggr[r] += attn * msg   (thread = one edge x 4 channels)
// ---------------------------------------------------------------------------
__global__ void cbf_aggregate_kernel(const unsigned short* __restrict__ gMsg,
                                     const float* __restrict__ gLogits,
                                     const int* __restrict__ receivers,
                                     const int* __restrict__ segMax,
                                     const float* __restrict__ denom,
                                     float* __restrict__ aggr, int nEdges) {
  int gid = blockIdx.x * blockDim.x + threadIdx.x;
  int total = nEdges * 32;
  if (gid >= total) return;
  int e = gid >> 5, q = gid & 31;
  int r = receivers[e];
  float attn = __expf(gLogits[e] - orderedDec(segMax[r])) / (denom[r] + 1e-9f);
  const unsigned short* mp = gMsg + (size_t)e * MSGD + q * 4;
#pragma unroll
  for (int i = 0; i < 4; ++i)
    atomicAdd(&aggr[(size_t)r * MSGD + q * 4 + i], attn * bf2f(mp[i]));
}

// ---------------------------------------------------------------------------
// K4: agent MLP 128 -> 256 -> 256 -> tanh(1) via WMMA
// ---------------------------------------------------------------------------
#define K4_PER_WAVE ((16*MSGD + 16*HIDD + 16*HIDD)*2)   // 20480 B
#define K4_SMEM (HIDD*MSGD*2 + HIDD*HIDD*2 + 3*HIDD*4 + 4*K4_PER_WAVE)

__global__ void __launch_bounds__(128)
cbf_agent_mlp_kernel(const float* __restrict__ aggr,
                     const float* __restrict__ Wh1, const float* __restrict__ Bh1,
                     const float* __restrict__ Wh2, const float* __restrict__ Bh2,
                     const float* __restrict__ Wout, const float* __restrict__ bOut,
                     int nAgents, float* __restrict__ out) {
  extern __shared__ char smem[];
  unsigned short* sW1 = (unsigned short*)smem;       // [256][128] bf16
  unsigned short* sW2 = sW1 + HIDD * MSGD;           // [256][256] bf16
  float* sWo = (float*)(sW2 + HIDD * HIDD);          // 256
  float* sB1 = sWo + HIDD;
  float* sB2 = sB1 + HIDD;
  char*  pw  = (char*)(sB2 + HIDD);

  const int wave = threadIdx.x >> 5, lane = threadIdx.x & 31;
  unsigned short* sX  = (unsigned short*)(pw + wave * K4_PER_WAVE); // [16][128]
  unsigned short* sH1 = sX + 16 * MSGD;                             // [16][256]
  unsigned short* sH2 = sH1 + 16 * HIDD;                            // [16][256]

  for (int i = threadIdx.x; i < MSGD * HIDD; i += blockDim.x) {
    int k = i / HIDD, n = i - k * HIDD;
    sW1[n * MSGD + k] = f2bf(Wh1[i]);
  }
  for (int i = threadIdx.x; i < HIDD * HIDD; i += blockDim.x) {
    int k = i / HIDD, n = i - k * HIDD;
    sW2[n * HIDD + k] = f2bf(Wh2[i]);
  }
  for (int i = threadIdx.x; i < HIDD; i += blockDim.x) {
    sWo[i] = Wout[i]; sB1[i] = Bh1[i]; sB2[i] = Bh2[i];
  }
  __syncthreads();

  const int col = lane & 15, half = lane >> 4;
  const int nTiles = (nAgents + 15) >> 4;

  for (int tile = blockIdx.x * 4 + wave; tile < nTiles; tile += gridDim.x * 4) {
    const int a0 = tile << 4;

    for (int idx = lane; idx < 16 * MSGD; idx += 32) {
      int r = idx / MSGD, c = idx - r * MSGD;
      int a = a0 + r;
      sX[idx] = f2bf((a < nAgents) ? aggr[(size_t)a * MSGD + c] : 0.f);
    }
    wave_lds_fence();

    for (int nt = 0; nt < 16; ++nt) {
      v8f acc = {};
#pragma unroll
      for (int kt = 0; kt < 4; ++kt)
        acc = wmma_bf16(fragA(sX, MSGD, kt * 32, lane),
                        fragB(sW1, MSGD, nt * 16, kt * 32, lane), acc);
      float bias = sB1[nt * 16 + col];
#pragma unroll
      for (int r = 0; r < 8; ++r) {
        float v = acc[r] + bias; v = v > 0.f ? v : 0.f;
        sH1[(r + 8 * half) * HIDD + nt * 16 + col] = f2bf(v);
      }
    }
    wave_lds_fence();

    for (int nt = 0; nt < 16; ++nt) {
      v8f acc = {};
#pragma unroll
      for (int kt = 0; kt < 8; ++kt)
        acc = wmma_bf16(fragA(sH1, HIDD, kt * 32, lane),
                        fragB(sW2, HIDD, nt * 16, kt * 32, lane), acc);
      float bias = sB2[nt * 16 + col];
#pragma unroll
      for (int r = 0; r < 8; ++r) {
        float v = acc[r] + bias; v = v > 0.f ? v : 0.f;
        sH2[(r + 8 * half) * HIDD + nt * 16 + col] = f2bf(v);
      }
    }
    wave_lds_fence();

    if (lane < 16) {
      int a = a0 + lane;
      if (a < nAgents) {
        float acc = bOut[0];
#pragma unroll 4
        for (int c = 0; c < HIDD; ++c) acc += bf2f(sH2[lane * HIDD + c]) * sWo[c];
        out[a] = tanhf(acc);
      }
    }
  }
}

// ---------------------------------------------------------------------------
// Host-side launcher
// ---------------------------------------------------------------------------
extern "C" void kernel_launch(void* const* d_in, const int* in_sizes, int n_in,
                              void* d_out, int out_size, void* d_ws, size_t ws_size,
                              hipStream_t stream) {
  const float* nodeF     = (const float*)d_in[0];
  const float* edgeF     = (const float*)d_in[1];
  const float* W1        = (const float*)d_in[2];
  const float* B1        = (const float*)d_in[3];
  const float* W2        = (const float*)d_in[4];
  const float* B2        = (const float*)d_in[5];
  const float* wGate     = (const float*)d_in[6];
  const float* bGate     = (const float*)d_in[7];
  const float* Wh1       = (const float*)d_in[8];
  const float* Bh1       = (const float*)d_in[9];
  const float* Wh2       = (const float*)d_in[10];
  const float* Bh2       = (const float*)d_in[11];
  const float* Wout      = (const float*)d_in[12];
  const float* bOut      = (const float*)d_in[13];
  const int*   senders   = (const int*)d_in[14];
  const int*   receivers = (const int*)d_in[15];

  const int nEdges  = in_sizes[14];
  const int nNodes  = in_sizes[0] / ND;
  const int nAgents = out_size;           // reference output = (n_agents, 1) f32

  // Workspace carve (256B aligned)
  char* ws = (char*)d_ws;
  size_t p = 0;
  auto carve = [&](size_t bytes) { size_t o = p; p = (p + bytes + 255) & ~(size_t)255; return o; };
  int*            segMax = (int*)           (ws + carve((size_t)nNodes * 4));
  float*          denom  = (float*)         (ws + carve((size_t)nNodes * 4));
  float*          aggr   = (float*)         (ws + carve((size_t)nNodes * MSGD * 4));
  float*          logits = (float*)         (ws + carve((size_t)nEdges * 4));
  unsigned short* msg    = (unsigned short*)(ws + carve((size_t)nEdges * MSGD * 2));
  (void)ws_size; (void)n_in;

  cbf_init_kernel<<<2048, 256, 0, stream>>>(segMax, denom, aggr, nNodes);

  cbf_edge_mlp_kernel<<<2048, 128, (size_t)K1_SMEM, stream>>>(
      nodeF, edgeF, W1, B1, W2, B2, wGate, bGate, senders, receivers,
      nEdges, msg, logits, segMax);

  cbf_denom_kernel<<<(nEdges + 255) / 256, 256, 0, stream>>>(
      logits, receivers, segMax, denom, nEdges);

  cbf_aggregate_kernel<<<(nEdges * 32 + 255) / 256, 256, 0, stream>>>(
      msg, logits, receivers, segMax, denom, aggr, nEdges);

  int tiles = (nAgents + 15) / 16;
  cbf_agent_mlp_kernel<<<(tiles + 3) / 4, 128, (size_t)K4_SMEM, stream>>>(
      aggr, Wh1, Bh1, Wh2, Bh2, Wout, bOut, nAgents, (float*)d_out);
}
